// Encoder_52450140619229
// MI455X (gfx1250) — compile-verified, matrix operand order checked by reading, not run on hardware
//
#include <hip/hip_runtime.h>

// ---------------------------------------------------------------------------
// GRU encoder for MI455X (gfx1250, wave32, WMMA, TDM).
//   T=512, B=128, V=50000, E=256, U=256
// Plan:
//   K0: transpose+convert weights to bf16 ([N][K] layout for WMMA B-frags)
//   K1: embedding gather -> X bf16 [T*B][E]
//   K2: parallel input projections Xg/Xc, stored in WMMA C-fragment-major
//       layout ([tile][lane][8] bf16) so producer stores and consumer loads
//       are single b128 per lane.
//   K3: persistent single-workgroup recurrent scan (32 waves, h in LDS,
//       recurrent weights L2-resident, 2 barriers/step, bf16 WMMA f32-accum,
//       TDM double-buffered staging of Xc fragments into LDS).
// ---------------------------------------------------------------------------

#define TT 512
#define BB 128
#define VV 50000
#define EE 256
#define UU 256
#define GG 512   // 2*U gate width

typedef __attribute__((ext_vector_type(16))) __bf16 v16bf;
typedef __attribute__((ext_vector_type(8)))  float  v8f;
typedef unsigned int v4u __attribute__((ext_vector_type(4)));
typedef int          v8i __attribute__((ext_vector_type(8)));
typedef int          v4i __attribute__((ext_vector_type(4)));

union FragBF {               // 16x32 bf16 A / 32x16 bf16 B operand fragment
    v16bf v;
    uint4 q[2];
    unsigned short s[16];
};
union Frag8 {                // one lane's 8 C-fragment values as bf16
    uint4 q;
    unsigned short s[8];
};

__device__ __forceinline__ unsigned short f2bf(float f) {
    unsigned int x = __float_as_uint(f);
    unsigned int r = (x + 0x7FFFu + ((x >> 16) & 1u)) >> 16;  // RNE
    return (unsigned short)r;
}
__device__ __forceinline__ float bf2f(unsigned short h) {
    return __uint_as_float(((unsigned int)h) << 16);
}
__device__ __forceinline__ float sigmoidf_(float x) {
    return 1.0f / (1.0f + __expf(-x));
}

// ---------------------------------------------------------------------------
// K0: weight transpose + bf16 convert ([N][K] row-major for B fragments).
// ---------------------------------------------------------------------------
__global__ __launch_bounds__(256) void weights_prep_kernel(
    const float* __restrict__ W_g, const float* __restrict__ W_c,
    unsigned short* __restrict__ WgxT, unsigned short* __restrict__ WghT,
    unsigned short* __restrict__ WcxT, unsigned short* __restrict__ WchT)
{
    int idx = blockIdx.x * blockDim.x + threadIdx.x;
    const int NGX = GG * EE, NGH = GG * UU, NCX = UU * EE, NCH = UU * UU;
    if (idx < NGX) {
        int n = idx / EE, k = idx % EE;
        WgxT[idx] = f2bf(W_g[(size_t)k * GG + n]);
    } else if (idx < NGX + NGH) {
        int j = idx - NGX;
        int n = j / UU, k = j % UU;
        WghT[j] = f2bf(W_g[(size_t)(EE + k) * GG + n]);
    } else if (idx < NGX + NGH + NCX) {
        int j = idx - NGX - NGH;
        int n = j / EE, k = j % EE;
        WcxT[j] = f2bf(W_c[(size_t)k * UU + n]);
    } else if (idx < NGX + NGH + NCX + NCH) {
        int j = idx - NGX - NGH - NCX;
        int n = j / UU, k = j % UU;
        WchT[j] = f2bf(W_c[(size_t)(EE + k) * UU + n]);
    }
}

// ---------------------------------------------------------------------------
// K1: embedding gather -> bf16 X [T*B][E]
// ---------------------------------------------------------------------------
__global__ __launch_bounds__(256) void embed_gather_kernel(
    const int* __restrict__ inputs, const float* __restrict__ emb,
    unsigned short* __restrict__ X)
{
    size_t idx = (size_t)blockIdx.x * blockDim.x + threadIdx.x;
    if (idx >= (size_t)TT * BB * EE) return;
    size_t row = idx >> 8;
    int    e   = (int)(idx & (EE - 1));
    int    tok = inputs[row];
    X[idx] = f2bf(emb[(size_t)tok * EE + e]);
}

// ---------------------------------------------------------------------------
// A/B operand fragment load (row-major bf16 src [*, ld], tile K-offset k0).
// Lanes 0-15: K {0..7, 16..23}; lanes 16-31: K {8..15, 24..31}.
// ---------------------------------------------------------------------------
__device__ __forceinline__ void load_frag(FragBF& f, const unsigned short* base,
                                          int row, int ld, int k0, int hl) {
    const unsigned short* p = base + (size_t)row * ld + k0 + hl * 8;
    f.q[0] = *(const uint4*)(p);
    f.q[1] = *(const uint4*)(p + 16);
}

// ---------------------------------------------------------------------------
// K2: input projections. One wave per 16x16 output tile; bias folded.
// Output layout is C-fragment-major:
//   XgF[(mt*32 + nt)*256 + lane*8 + j]   (gate tiles,     nt 0..31)
//   XcF[(mt*16 + nt)*256 + lane*8 + j]   (candidate tiles, nt 0..15)
// ---------------------------------------------------------------------------
__global__ __launch_bounds__(256) void input_proj_kernel(
    const unsigned short* __restrict__ X,
    const unsigned short* __restrict__ WgxT,
    const unsigned short* __restrict__ WcxT,
    const float* __restrict__ b_g, const float* __restrict__ b_c,
    unsigned short* __restrict__ XgF, unsigned short* __restrict__ XcF)
{
    const int NT = 48;                       // 32 gate + 16 candidate n-tiles
    size_t waveId = ((size_t)blockIdx.x * blockDim.x + threadIdx.x) >> 5;
    size_t nTiles = (size_t)(TT * BB / 16) * NT;
    if (waveId >= nTiles) return;

    int lane = threadIdx.x & 31;
    int hl   = lane >> 4;
    int l15  = lane & 15;

    int m_tile = (int)(waveId / NT);         // 0..4095
    int nt     = (int)(waveId % NT);
    int m0     = m_tile * 16;

    bool isG = nt < 32;
    int  ntc = isG ? nt : (nt - 32);
    int  n0  = ntc * 16;
    const unsigned short* WT = isG ? WgxT : WcxT;

    v8f acc = {};
    #pragma unroll
    for (int k = 0; k < EE; k += 32) {
        FragBF a, b;
        load_frag(a, X,  m0 + l15, EE, k, hl);
        load_frag(b, WT, n0 + l15, EE, k, hl);
        acc = __builtin_amdgcn_wmma_f32_16x16x32_bf16(
                  false, a.v, false, b.v, (short)0, acc, false, false);
    }

    float bias = isG ? b_g[n0 + l15] : b_c[n0 + l15];
    Frag8 st;
    #pragma unroll
    for (int j = 0; j < 8; ++j) st.s[j] = f2bf(acc[j] + bias);

    if (isG) *(uint4*)(XgF + ((size_t)m_tile * 32 + ntc) * 256 + lane * 8) = st.q;
    else     *(uint4*)(XcF + ((size_t)m_tile * 16 + ntc) * 256 + lane * 8) = st.q;
}

// ---------------------------------------------------------------------------
// TDM: 1-D contiguous 64KB copy (one timestep of Xc fragments) into LDS.
// D# per CDNA5 ISA ch.8: group0 = {count, lds_addr, global_addr, type=2};
// group1 = {data_size=3 (8B), tensor_dim0=8192, tensor_dim1=1,
//           tile_dim0=8192, tile_dim1=1, dim0_stride=8192}; groups 2/3 = 0.
// This toolchain's builtin is the 6-arg form (clang-23 / therock-10.0):
//   (uint32x4 g0, int32x8 g1, int32x4 g2, int32x4 g3, int32x8, i32 cpol)
// ---------------------------------------------------------------------------
#if __has_builtin(__builtin_amdgcn_tensor_load_to_lds)
#define USE_TDM 1
#else
#define USE_TDM 0
#endif

#define LDS_H_OFF    0u
#define LDS_RH_OFF   65536u
#define LDS_XC0_OFF  131072u
#define LDS_XC1_OFF  196608u
#define LDS_LEN_OFF  262144u
#define LDS_TOTAL    (262144u + BB * 4u)

#if USE_TDM
__device__ __forceinline__ void tdm_stage_xc(const unsigned short* XcF,
                                             int step, unsigned lds_off) {
    unsigned long long ga =
        (unsigned long long)(uintptr_t)(XcF + (size_t)step * (BB * UU));
    const unsigned TD0 = 8192u;   // 8-byte elements per step chunk
    v4u g0;
    g0.x = 1u;                                              // count=1 (valid)
    g0.y = lds_off;                                         // lds_addr (bytes)
    g0.z = (unsigned)(ga & 0xFFFFFFFFu);                    // global_addr lo
    g0.w = (unsigned)((ga >> 32) & 0x01FFFFFFu) | (2u << 30); // addr hi | type=2
    v8i g1;
    g1[0] = (int)(3u << 16);                                // data_size=3 (8B)
    g1[1] = (int)((TD0 & 0xFFFFu) << 16);                   // tensor_dim0 lo16
    g1[2] = (int)(((TD0 >> 16) & 0xFFFFu) | (1u << 16));    // dim0 hi | dim1=1 lo
    g1[3] = (int)((TD0 & 0xFFFFu) << 16);                   // dim1 hi=0 | tile_dim0
    g1[4] = (int)1u;                                        // tile_dim1=1, tile_dim2=0
    g1[5] = (int)TD0;                                       // dim0_stride lo32
    g1[6] = 0;                                              // stride hi / dim1_stride
    g1[7] = 0;
    v4i z4 = {0, 0, 0, 0};
    v8i z8 = {0, 0, 0, 0, 0, 0, 0, 0};
    __builtin_amdgcn_tensor_load_to_lds(g0, g1, z4, z4, z8, 0);
}
#endif

// ---------------------------------------------------------------------------
// K3: persistent recurrent scan. One workgroup, 32 wave32s on one WGP.
//  LDS: h bf16 (64KB) + r*h bf16 (64KB) + Xc double-buffer (2x64KB) + lens.
//  Wave w: m_tile = w>>2, n-group = w&3.
//   Phase A: 4 r-tiles + matching 4 u-tiles: acc = XgF + h @ WghT, sigmoid;
//            r*h -> LDS, u stays in VGPRs.
//   Phase B: 4 candidate tiles: acc = staged Xc + rh @ WchT, tanh; combine
//            with u and h, apply dynamic_rnn length mask, write y, update h.
//  TDM for step t+1 is issued at the top of step t; its s_wait_tensorcnt sits
//  just before the end-of-step barrier, hiding the copy behind ~96 WMMA/wave.
// ---------------------------------------------------------------------------
__global__ __launch_bounds__(1024) void gru_scan_kernel(
    const unsigned short* __restrict__ XgF,   // fragment-major, bias folded
    const unsigned short* __restrict__ XcF,   // fragment-major, bias folded
    const unsigned short* __restrict__ WghT,  // [512][256] bf16
    const unsigned short* __restrict__ WchT,  // [256][256] bf16
    const int*            __restrict__ lengths,
    float*                __restrict__ out)   // [T*B*U] outputs ++ [B*U] h_final
{
    extern __shared__ char smem[];
    unsigned short* h_s   = (unsigned short*)(smem + LDS_H_OFF);
    unsigned short* rh_s  = (unsigned short*)(smem + LDS_RH_OFF);
    int*            len_s = (int*)(smem + LDS_LEN_OFF);

    const int tid  = threadIdx.x;
    const int wave = tid >> 5;
    const int lane = tid & 31;
    const int hl   = lane >> 4;
    const int l15  = lane & 15;
    const int m_tile = wave >> 2;   // 0..7
    const int ng     = wave & 3;    // 0..3
    const int m0     = m_tile * 16;

    for (int i = tid; i < BB * UU; i += 1024) h_s[i] = 0;
    if (tid < BB) len_s[tid] = lengths[tid];
#if USE_TDM
    if (wave == 0) {                       // stage step 0 synchronously
        tdm_stage_xc(XcF, 0, LDS_XC0_OFF);
        __builtin_amdgcn_s_wait_tensorcnt(0);
    }
#endif
    __syncthreads();

    for (int t = 0; t < TT; ++t) {
#if USE_TDM
        if (wave == 0 && t + 1 < TT)       // overlap next step's copy
            tdm_stage_xc(XcF, t + 1, ((t + 1) & 1) ? LDS_XC1_OFF : LDS_XC0_OFF);
        const unsigned short* xcs =
            (const unsigned short*)(smem + ((t & 1) ? LDS_XC1_OFF : LDS_XC0_OFF));
#endif
        if (t + 1 < TT)
            __builtin_prefetch(XgF + (size_t)(t + 1) * (BB * GG), 0, 1);

        v8f uacc[4];

        // ---------------- Phase A: gates -----------------------------------
        #pragma unroll
        for (int i = 0; i < 4; ++i) {
            #pragma unroll
            for (int which = 0; which < 2; ++which) {
                int n_tile = which ? (16 + ng * 4 + i) : (ng * 4 + i);
                int n0 = n_tile * 16;
                int n  = n0 + l15;

                Frag8 cin;
                cin.q = *(const uint4*)(XgF +
                          (((size_t)(t * 8 + m_tile) * 32 + n_tile) * 256) + lane * 8);
                v8f acc;
                #pragma unroll
                for (int j = 0; j < 8; ++j) acc[j] = bf2f(cin.s[j]);

                #pragma unroll
                for (int k = 0; k < UU; k += 32) {
                    FragBF a, b;
                    load_frag(a, h_s,  m0 + l15, UU, k, hl);   // ds_load_b128
                    load_frag(b, WghT, n0 + l15, UU, k, hl);   // L2-resident
                    acc = __builtin_amdgcn_wmma_f32_16x16x32_bf16(
                              false, a.v, false, b.v, (short)0, acc, false, false);
                }
                if (which == 0) {
                    #pragma unroll
                    for (int j = 0; j < 8; ++j) {
                        int m = m0 + hl * 8 + j;
                        float r = sigmoidf_(acc[j]);
                        float h = bf2f(h_s[m * UU + n]);
                        rh_s[m * UU + n] = f2bf(r * h);
                    }
                } else {
                    #pragma unroll
                    for (int j = 0; j < 8; ++j) uacc[i][j] = sigmoidf_(acc[j]);
                }
            }
        }
        __syncthreads();   // rh complete before candidate GEMM

        // ---------------- Phase B: candidate + state update -----------------
        #pragma unroll
        for (int i = 0; i < 4; ++i) {
            int n_tile = ng * 4 + i;
            int n0 = n_tile * 16;
            int n  = n0 + l15;

            Frag8 cin;
#if USE_TDM
            cin.q = *(const uint4*)(xcs +
                      ((size_t)(m_tile * 16 + n_tile) * 256) + lane * 8);
#else
            cin.q = *(const uint4*)(XcF +
                      (((size_t)(t * 8 + m_tile) * 16 + n_tile) * 256) + lane * 8);
#endif
            v8f acc;
            #pragma unroll
            for (int j = 0; j < 8; ++j) acc[j] = bf2f(cin.s[j]);

            #pragma unroll
            for (int k = 0; k < UU; k += 32) {
                FragBF a, b;
                load_frag(a, rh_s, m0 + l15, UU, k, hl);
                load_frag(b, WchT, n0 + l15, UU, k, hl);
                acc = __builtin_amdgcn_wmma_f32_16x16x32_bf16(
                          false, a.v, false, b.v, (short)0, acc, false, false);
            }
            #pragma unroll
            for (int j = 0; j < 8; ++j) {
                int m = m0 + hl * 8 + j;
                float c  = tanhf(acc[j]);
                float u  = uacc[i][j];
                float h  = bf2f(h_s[m * UU + n]);
                float hn = u * h + (1.0f - u) * c;
                bool valid = t < len_s[m];
                out[(size_t)(t * BB + m) * UU + n] = valid ? hn : 0.0f;
                h_s[m * UU + n] = f2bf(valid ? hn : h);
            }
        }
#if USE_TDM
        if (wave == 0) __builtin_amdgcn_s_wait_tensorcnt(0);
#endif
        __syncthreads();   // h stable + next Xc staged before next step
    }

    // h_final
    for (int i = tid; i < BB * UU; i += 1024)
        out[(size_t)TT * BB * UU + i] = bf2f(h_s[i]);
}

// ---------------------------------------------------------------------------
// Host-side launch. Workspace layout (bytes):
//   X    bf16 T*B*E   @ 0            (33,554,432)
//   XgF  bf16 T*B*2U  @ 33,554,432   (67,108,864)   fragment-major
//   XcF  bf16 T*B*U   @ 100,663,296  (33,554,432)   fragment-major
//   WgxT bf16 512*256 @ 134,217,728  (   262,144)
//   WghT bf16 512*256 @ 134,479,872  (   262,144)
//   WcxT bf16 256*256 @ 134,742,016  (   131,072)
//   WchT bf16 256*256 @ 134,873,088  (   131,072)   total ~128.8 MiB
// ---------------------------------------------------------------------------
extern "C" void kernel_launch(void* const* d_in, const int* in_sizes, int n_in,
                              void* d_out, int out_size, void* d_ws, size_t ws_size,
                              hipStream_t stream)
{
    const int*   inputs  = (const int*)  d_in[0];
    const int*   lengths = (const int*)  d_in[1];
    const float* emb     = (const float*)d_in[2];
    const float* W_g     = (const float*)d_in[3];
    const float* b_g     = (const float*)d_in[4];
    const float* W_c     = (const float*)d_in[5];
    const float* b_c     = (const float*)d_in[6];
    float*       out     = (float*)d_out;

    char* ws = (char*)d_ws;
    unsigned short* X    = (unsigned short*)(ws);
    unsigned short* XgF  = (unsigned short*)(ws + 33554432ull);
    unsigned short* XcF  = (unsigned short*)(ws + 100663296ull);
    unsigned short* WgxT = (unsigned short*)(ws + 134217728ull);
    unsigned short* WghT = (unsigned short*)(ws + 134479872ull);
    unsigned short* WcxT = (unsigned short*)(ws + 134742016ull);
    unsigned short* WchT = (unsigned short*)(ws + 134873088ull);

    weights_prep_kernel<<<(GG*EE*2 + UU*EE + UU*UU + 255) / 256, 256, 0, stream>>>(
        W_g, W_c, WgxT, WghT, WcxT, WchT);

    {
        size_t n = (size_t)TT * BB * EE;
        embed_gather_kernel<<<(unsigned)((n + 255) / 256), 256, 0, stream>>>(
            inputs, emb, X);
    }

    {
        size_t waves  = (size_t)(TT * BB / 16) * 48;
        size_t blocks = (waves * 32 + 255) / 256;
        input_proj_kernel<<<(unsigned)blocks, 256, 0, stream>>>(
            X, WgxT, WcxT, b_g, b_c, XgF, XcF);
    }

    gru_scan_kernel<<<1, 1024, LDS_TOTAL, stream>>>(
        XgF, XcF, WghT, WchT, lengths, out);
}
